// CustomizeLSTMCell_76536317215121
// MI455X (gfx1250) — matrix operand order repaired and checked by reading, not compiled, not run on hardware
//
#include <hip/hip_runtime.h>
#include <hip/hip_bf16.h>

// ---------------------------------------------------------------------------
// Fused 4-GEMM LSTM gate kernel for gfx1250 (MI455X, wave32, WMMA).
//   pre = [s_in|s_out|h_in|h_out] ([N,600]) @ [W_in;W_out;U_in;U_out] ([600,150])
//   gate = sigmoid(pre); cell = gate*last_c + gate^2; hidden = gate*tanh(cell)
// Memory-bound (AI ~43 FLOP/B -> ~18us at 23.3 TB/s). Math via
// v_wmma_f32_16x16x32_f16; the whole transposed/padded f16 weight matrix
// (202 KB) lives in LDS (320 KB/WGP), so HBM traffic is activations only.
// ---------------------------------------------------------------------------

#define UNITS 150
#define KPAD  160                 // per-source-matrix padded K (5 x 32)
#define NMATS 4
#define KROW  648                 // LDS row stride in halves (bank-conflict pad)
#define NPAD  160                 // padded N (10 x 16)
#define NTILES 10
#define WAVES_PER_BLOCK 16
#define THREADS (WAVES_PER_BLOCK * 32)
#define LDS_HALVES (NPAD * KROW)  // 103,680 halves = 207,360 bytes

typedef __attribute__((ext_vector_type(16))) _Float16 v16h;
typedef __attribute__((ext_vector_type(8)))  _Float16 v8h;
typedef __attribute__((ext_vector_type(8)))  float    v8f;

static __device__ __forceinline__ v16h load_b_frag(const _Float16* bbase, int nt) {
    // lane's B fragment: column nt*16 + (lane&15), K 8h..8h+7 & 16+8h..16+8h+7
    const _Float16* bp = bbase + nt * (16 * KROW);
    const v8h lo = *(const v8h*)bp;
    const v8h hi = *(const v8h*)(bp + 16);
    return __builtin_shufflevector(lo, hi, 0,1,2,3,4,5,6,7,8,9,10,11,12,13,14,15);
}

// Unguarded A fragment: rows are >= 8B aligned, all K in range.
static __device__ __forceinline__ v16h load_a_frag(const float* src, int kbase) {
    v16h a;
#pragma unroll
    for (int p = 0; p < 4; ++p) {
        const float2 g0 = *(const float2*)(src + kbase + 2 * p);
        const float2 g1 = *(const float2*)(src + kbase + 16 + 2 * p);
        a[2 * p]         = (_Float16)g0.x;
        a[2 * p + 1]     = (_Float16)g0.y;
        a[8 + 2 * p]     = (_Float16)g1.x;
        a[8 + 2 * p + 1] = (_Float16)g1.y;
    }
    return a;
}

// Tail A fragment (k in [128,160)): group0 always valid, group1 branchless-masked.
static __device__ __forceinline__ v16h load_a_tail(const float* src, int kbase) {
    v16h a;
#pragma unroll
    for (int p = 0; p < 4; ++p) {
        const float2 g0 = *(const float2*)(src + kbase + 2 * p);   // <= 143: valid
        a[2 * p]     = (_Float16)g0.x;
        a[2 * p + 1] = (_Float16)g0.y;

        const int  k1 = kbase + 16 + 2 * p;                        // 144..158
        const bool ok = (k1 <= UNITS - 2);                         // pair fits
        const float2 g1 = *(const float2*)(src + (ok ? k1 : 0));   // clamped, safe
        const float  m  = ok ? 1.f : 0.f;                          // v_cndmask
        a[8 + 2 * p]     = (_Float16)(g1.x * m);
        a[8 + 2 * p + 1] = (_Float16)(g1.y * m);
    }
    return a;
}

// One K-step: 10 WMMAs against double-buffered B fragments from LDS.
static __device__ __forceinline__ void kstep(v8f acc[NTILES], v16h a,
                                             const _Float16* bbase) {
    v16h b0 = load_b_frag(bbase, 0);
    v16h b1 = load_b_frag(bbase, 1);
#pragma unroll
    for (int nt = 0; nt < NTILES; nt += 2) {
        acc[nt] = __builtin_amdgcn_wmma_f32_16x16x32_f16(
            false, a, false, b0, (short)0, acc[nt], false, false);
        if (nt + 2 < NTILES) b0 = load_b_frag(bbase, nt + 2);
        acc[nt + 1] = __builtin_amdgcn_wmma_f32_16x16x32_f16(
            false, a, false, b1, (short)0, acc[nt + 1], false, false);
        if (nt + 3 < NTILES) b1 = load_b_frag(bbase, nt + 3);
    }
}

__global__ __launch_bounds__(THREADS, 1)
void lstm_fused_wmma(const float* __restrict__ s_in,
                     const float* __restrict__ s_out,
                     const float* __restrict__ h_in,
                     const float* __restrict__ h_out,
                     const float* __restrict__ last_c,
                     const float* __restrict__ w_in,
                     const float* __restrict__ w_out,
                     const float* __restrict__ u_in,
                     const float* __restrict__ u_out,
                     float* __restrict__ out_h,
                     float* __restrict__ out_c,
                     int n_tokens, int n_tiles)
{
    extern __shared__ _Float16 sB[];   // B^T, f16, [NPAD][KROW], zero padded

    const int tid = threadIdx.x;

    // ---- stage B^T (all four weight matrices) into LDS as f16 ------------
    {
        unsigned* z = (unsigned*)sB;
        for (int i = tid; i < LDS_HALVES / 2; i += THREADS) z[i] = 0u;
    }
    __syncthreads();
    {
        const float* wmats[NMATS] = { w_in, w_out, u_in, u_out };
#pragma unroll
        for (int mat = 0; mat < NMATS; ++mat) {
            const float* w = wmats[mat];
            for (int i = tid; i < UNITS * UNITS; i += THREADS) {
                const int k = i / UNITS;
                const int n = i - k * UNITS;
                sB[n * KROW + mat * KPAD + k] = (_Float16)w[i];  // transpose
            }
        }
    }
    __syncthreads();

    const int wave = tid >> 5;
    const int lane = tid & 31;
    const int half = lane >> 4;     // 0: lanes 0-15, 1: lanes 16-31
    const int l16  = lane & 15;

    const float* amats[NMATS] = { s_in, s_out, h_in, h_out };

    for (int tile = blockIdx.x * WAVES_PER_BLOCK + wave; tile < n_tiles;
         tile += gridDim.x * WAVES_PER_BLOCK) {

        const int mbase = tile * 16;
        int arow = mbase + l16;                    // A row owned by this lane
        if (arow >= n_tokens) arow = n_tokens - 1; // clamp (100000 % 16 == 0)

        v8f acc[NTILES] = {};

        const int kb = half * 8;   // f16 A layout: lane<16 -> K 0..7 & 16..23
        const _Float16* brow = &sB[l16 * KROW + half * 8];

#pragma unroll
        for (int mat = 0; mat < NMATS; ++mat) {
            const float* src = amats[mat] + (size_t)arow * UNITS;
            const _Float16* bmat = brow + mat * KPAD;
#pragma unroll 1
            for (int sub = 0; sub < 4; ++sub) {            // k 0..127: no guards
                const v16h a = load_a_frag(src, sub * 32 + kb);
                kstep(acc, a, bmat + sub * 32);
            }
            {                                              // k 128..159: tail
                const v16h a = load_a_tail(src, 128 + kb);
                kstep(acc, a, bmat + 128);
            }
        }

        // ---- epilogue: FULLY UNROLLED so acc[][] stays in registers -------
        // (a rolled loop would dynamically index the acc array -> scratch)
#pragma unroll
        for (int nt = 0; nt < NTILES; ++nt) {
            const int col = nt * 16 + l16;       // nt<=8: provably < 150 (folds)
            if (col < UNITS) {
                const float* lc_col = last_c + col;
                float*       oh_col = out_h  + col;
                float*       oc_col = out_c  + col;
#pragma unroll
                for (int v = 0; v < 8; ++v) {
                    const int row = mbase + v + half * 8;
                    if (row < n_tokens) {
                        const size_t off = (size_t)row * UNITS;
                        const float pre = acc[nt][v];
                        const float g   = 1.f / (1.f + __expf(-pre));
                        const float cs  = g * lc_col[off] + g * g;
                        // tanh(x) = 1 - 2/(exp(2x)+1); |cs| bounded, no overflow
                        const float th  = 1.f - 2.f / (__expf(2.f * cs) + 1.f);
                        oh_col[off] = g * th;
                        oc_col[off] = cs;
                    }
                }
            }
        }
    }
}

extern "C" void kernel_launch(void* const* d_in, const int* in_sizes, int n_in,
                              void* d_out, int out_size, void* d_ws, size_t ws_size,
                              hipStream_t stream) {
    (void)n_in; (void)out_size; (void)d_ws; (void)ws_size;

    const float* s_in   = (const float*)d_in[0];
    const float* s_out  = (const float*)d_in[1];
    const float* h_in   = (const float*)d_in[2];
    const float* h_out  = (const float*)d_in[3];
    const float* last_c = (const float*)d_in[4];
    const float* w_in   = (const float*)d_in[5];
    const float* w_out  = (const float*)d_in[6];
    const float* u_in   = (const float*)d_in[7];
    const float* u_out  = (const float*)d_in[8];

    const int n_tokens = in_sizes[0] / UNITS;          // 100000
    const int n_tiles  = (n_tokens + 15) / 16;         // 6250

    float* out_h = (float*)d_out;
    float* out_c = out_h + (size_t)n_tokens * UNITS;   // tuple order: hidden, cell

    int blocks = (n_tiles + WAVES_PER_BLOCK - 1) / WAVES_PER_BLOCK;  // 391
    if (blocks > 1024) blocks = 1024;

    const size_t lds_bytes = (size_t)LDS_HALVES * sizeof(_Float16);  // 207,360 B

    lstm_fused_wmma<<<blocks, THREADS, lds_bytes, stream>>>(
        s_in, s_out, h_in, h_out, last_c,
        w_in, w_out, u_in, u_out,
        out_h, out_c, n_tokens, n_tiles);
}
